// TransformerLMHead_54812372631859
// MI455X (gfx1250) — matrix operand order repaired and checked by reading, not compiled, not run on hardware
//
#include <hip/hip_runtime.h>
#include <hip/hip_bf16.h>
#include <math.h>

// ---------------------------------------------------------------------------
// Shapes (compile-time, from the reference)
// ---------------------------------------------------------------------------
#define SQ  2048   // sequence length S
#define DD  1024   // model dim D (== H*DK)
#define HHN 16     // heads
#define DKK 64     // head dim
#define DHH 4096   // MLP hidden
#define VVN 256    // vocab
#define LLN 4      // layers

typedef __bf16 bf16;
typedef __attribute__((ext_vector_type(16))) __bf16 bf16x16;
typedef __attribute__((ext_vector_type(8)))  float  f32x8;
typedef __attribute__((ext_vector_type(4)))  float  f32x4;
typedef __attribute__((ext_vector_type(4)))  unsigned int u32x4;
typedef __attribute__((ext_vector_type(8)))  int    i32x8;
typedef __attribute__((ext_vector_type(4)))  int    i32x4;

#if __has_builtin(__builtin_amdgcn_tensor_load_to_lds)
#define HAVE_TDM 1
#else
#define HAVE_TDM 0
#endif

// ---------------------------------------------------------------------------
// Fragment loaders (wave32 WMMA 16x16x32 bf16 layouts, CDNA5 ISA)
// A (16x32, MxK): lane<16 -> row=lane, K = {0..7, 16..23}; lane>=16 -> K={8..15,24..31}
// B (32x16, KxN): lane<16 -> col=lane, K = {0..7, 16..23}; lane>=16 -> K={8..15,24..31}
// ---------------------------------------------------------------------------
__device__ __forceinline__ bf16x16 load_fragA(const bf16* p) {
  union { bf16x16 v; f32x4 f[2]; } u;
  u.f[0] = *reinterpret_cast<const f32x4*>(p);
  u.f[1] = *reinterpret_cast<const f32x4*>(p + 16);
  return u.v;
}
__device__ __forceinline__ bf16x16 load_fragB(const bf16* p) {
  union { bf16x16 v; f32x4 f[2]; } u;
  u.f[0] = *reinterpret_cast<const f32x4*>(p);
  u.f[1] = *reinterpret_cast<const f32x4*>(p + 16);
  return u.v;
}
__device__ __forceinline__ f32x8 wmma_bf16(bf16x16 a, bf16x16 b, f32x8 c) {
  return __builtin_amdgcn_wmma_f32_16x16x32_bf16(false, a, false, b, (short)0, c,
                                                 false, false);
}

// ---------------------------------------------------------------------------
// CDNA5 async global->LDS copy (ASYNCcnt path, CDNA5 ISA)
// ---------------------------------------------------------------------------
__device__ __forceinline__ void async_copy_b128(void* lds_ptr, const void* gptr) {
  unsigned int lds = (unsigned int)(unsigned long long)lds_ptr;
  asm volatile("global_load_async_to_lds_b128 %0, %1, off"
               :: "v"(lds), "v"(gptr) : "memory");
}
#define WAIT_ASYNC(n) asm volatile("s_wait_asynccnt " #n ::: "memory")

// ---------------------------------------------------------------------------
// CDNA5 Tensor Data Mover: load a 2D tile (rows x 32 bf16, row stride = ld)
// from global memory into LDS.  D# layout per CDNA5 ISA.
// Issued by one wave; tracked by TENSORcnt.
// ---------------------------------------------------------------------------
#if HAVE_TDM
__device__ __forceinline__ void tdm_load_tile(const void* gptr, unsigned lds_off,
                                              int ld_elems, int rows) {
  const unsigned long long ga = (unsigned long long)gptr;
  u32x4 g0;
  g0.x = 1u;                                        // count=1, user descriptor
  g0.y = lds_off;                                   // lds_addr (bytes)
  g0.z = (unsigned)(ga & 0xffffffffu);              // global_addr[31:0]
  g0.w = (unsigned)((ga >> 32) & 0x01ffffffu)       // global_addr[56:32]
       | (2u << 30);                                // type=2 ("image")
  i32x8 g1;
  g1[0] = (int)(1u << 16);                          // wg_mask=0, data_size=1 (2B)
  g1[1] = (int)(32u << 16);                         // tensor_dim0[15:0]=32
  g1[2] = (int)((unsigned)rows << 16);              // tensor_dim0 hi=0 | tensor_dim1 lo
  g1[3] = (int)(32u << 16);                         // tensor_dim1 hi=0 | tile_dim0=32
  g1[4] = (int)(unsigned)rows;                      // tile_dim1=rows, tile_dim2=0
  g1[5] = (int)(unsigned)ld_elems;                  // tensor_dim0_stride[31:0]
  g1[6] = 0;                                        // stride hi, dim1_stride lo
  g1[7] = 0;
  const i32x4 gz = {0, 0, 0, 0};
#if __clang_major__ >= 23
  const i32x8 gz8 = {0, 0, 0, 0, 0, 0, 0, 0};
  __builtin_amdgcn_tensor_load_to_lds(g0, g1, gz, gz, gz8, 0);
#else
  __builtin_amdgcn_tensor_load_to_lds(g0, g1, gz, gz, 0);
#endif
}
#endif

// ---------------------------------------------------------------------------
// GEMM: C[M,N] = A[M,K](bf16) * Bt[N,K](bf16)^T + bias, fused epilogue.
// Block = 256 threads (8 waves) computing a 128x128 tile; each wave computes
// 32x64 (2x4 WMMA tiles).  Double-buffered LDS k-slabs:
//   A tile: TDM tensor_load_to_lds (TENSORcnt)  [fallback: async b128]
//   B tile: global_load_async_to_lds_b128 (ASYNCcnt)
// ACT: 0=none, 1=exact GELU.  OMODE: 0=f32 out, 1=bf16 out, 2=bf16 transposed.
// ---------------------------------------------------------------------------
template<int ACT, int OMODE>
__global__ __launch_bounds__(256) void gemm_wmma_kernel(
    const bf16* __restrict__ A, const bf16* __restrict__ Bt,
    const float* __restrict__ bias, float* __restrict__ outF,
    bf16* __restrict__ outB, int M, int N, int K, int lda, int ldb, int ldo)
{
  __shared__ __attribute__((aligned(16))) bf16 sA[2][128 * 32];
  __shared__ __attribute__((aligned(16))) bf16 sB[2][128 * 32];

  const int tid  = threadIdx.x;
  const int wave = tid >> 5;
  const int lane = tid & 31;
  const int half = lane >> 4;
  const int r15  = lane & 15;
  const int wm   = wave >> 1;          // 0..3 -> 32-row strip
  const int wn   = wave & 1;           // 0..1 -> 64-col strip
  const int bm   = blockIdx.y * 128;
  const int bn   = blockIdx.x * 128;

  // Per-thread staging of B (and A in fallback): 128 rows x 32 k = 512 x b128.
  const int c0   = tid * 2;
  const int srow = c0 >> 2;            // 0..127
  const int q0   = (c0 & 3) * 8;
  const int q1   = q0 + 8;
  const bf16* gB = Bt + (size_t)(bn + srow) * ldb;
  bf16* lB = &sB[0][srow * 32];
  const size_t bufStride = 128 * 32;

  auto issueB = [&](int buf, int k0) {
    async_copy_b128(lB + buf * bufStride + q0, gB + k0 + q0);
    async_copy_b128(lB + buf * bufStride + q1, gB + k0 + q1);
  };

#if HAVE_TDM
  const bf16* gAblk = A + (size_t)bm * lda;
  const unsigned ldsA0 = (unsigned)(unsigned long long)(void*)&sA[0][0];
  auto issueA = [&](int buf, int k0) {
    if (wave == 0)
      tdm_load_tile(gAblk + k0, ldsA0 + (unsigned)(buf * (int)bufStride * 2),
                    lda, 128);
  };
#define WAIT_ASYNC_NEXT WAIT_ASYNC(2)
#else
  const bf16* gA = A + (size_t)(bm + srow) * lda;
  bf16* lA = &sA[0][srow * 32];
  auto issueA = [&](int buf, int k0) {
    async_copy_b128(lA + buf * bufStride + q0, gA + k0 + q0);
    async_copy_b128(lA + buf * bufStride + q1, gA + k0 + q1);
  };
#define WAIT_ASYNC_NEXT WAIT_ASYNC(4)
#endif

  const f32x8 fz = {0.f, 0.f, 0.f, 0.f, 0.f, 0.f, 0.f, 0.f};
  f32x8 acc[2][4] = {{fz, fz, fz, fz}, {fz, fz, fz, fz}};

  issueA(0, 0);
  issueB(0, 0);
  const int nk = K >> 5;
  for (int ki = 0; ki < nk; ++ki) {
    const int cur = ki & 1;
    if (ki + 1 < nk) {
      issueA(cur ^ 1, (ki + 1) << 5);
      issueB(cur ^ 1, (ki + 1) << 5);
      WAIT_ASYNC_NEXT;                 // cur B slab landed (async loads in order)
#if HAVE_TDM
      if (wave == 0) __builtin_amdgcn_s_wait_tensorcnt(1);   // cur A slab landed
#endif
    } else {
      WAIT_ASYNC(0);
#if HAVE_TDM
      if (wave == 0) __builtin_amdgcn_s_wait_tensorcnt(0);
#endif
    }
    __syncthreads();

    bf16x16 af[2], bfr[4];
    af[0] = load_fragA(&sA[cur][(wm * 32 + r15) * 32 + half * 8]);
    af[1] = load_fragA(&sA[cur][(wm * 32 + 16 + r15) * 32 + half * 8]);
#pragma unroll
    for (int t = 0; t < 4; ++t)
      bfr[t] = load_fragB(&sB[cur][(wn * 64 + t * 16 + r15) * 32 + half * 8]);
#pragma unroll
    for (int mi = 0; mi < 2; ++mi)
#pragma unroll
      for (int t = 0; t < 4; ++t)
        acc[mi][t] = wmma_bf16(af[mi], bfr[t], acc[mi][t]);

    __syncthreads();                   // all reads done before next overwrite
  }

#pragma unroll
  for (int t = 0; t < 4; ++t) {
    const int col = bn + wn * 64 + t * 16 + r15;
    const float bc = bias ? bias[col] : 0.0f;
#pragma unroll
    for (int mi = 0; mi < 2; ++mi) {
#pragma unroll
      for (int r = 0; r < 8; ++r) {
        const int row = bm + wm * 32 + mi * 16 + r + half * 8;
        float v = acc[mi][t][r] + bc;
        if (ACT == 1) v = 0.5f * v * (1.0f + erff(v * 0.70710678118654752f));
        if (OMODE == 0)      outF[(size_t)row * ldo + col] = v;
        else if (OMODE == 1) outB[(size_t)row * ldo + col] = (bf16)v;
        else                 outB[(size_t)col * ldo + row] = (bf16)v;
      }
    }
  }
}

// ---------------------------------------------------------------------------
// Flash-style causal attention. grid=(S/16, H), block=32 (one wave).
// Q,K: [S, D] bf16 (head slice at h*64).  Vt: [H*DK, S] bf16 (pre-transposed).
// ---------------------------------------------------------------------------
__global__ __launch_bounds__(32) void attention_kernel(
    const bf16* __restrict__ Q, const bf16* __restrict__ Kmat,
    const bf16* __restrict__ Vt, bf16* __restrict__ O)
{
  const int qt = blockIdx.x, hh = blockIdx.y;
  const int lane = threadIdx.x;
  const int half = lane >> 4;
  const int r15  = lane & 15;
  const int q0   = qt * 16;

  __shared__ __attribute__((aligned(16))) bf16 pTile[16 * 32];

  const bf16* qbase = Q + (size_t)(q0 + r15) * DD + hh * DKK + half * 8;
  const bf16x16 aq0 = load_fragA(qbase);
  const bf16x16 aq1 = load_fragA(qbase + 32);

  f32x8 o0 = {0.f,0.f,0.f,0.f,0.f,0.f,0.f,0.f};
  f32x8 o1 = o0, o2 = o0, o3 = o0;
  float mrow[8], lrow[8];
#pragma unroll
  for (int r = 0; r < 8; ++r) { mrow[r] = -1e30f; lrow[r] = 0.f; }

  const int nkb = (q0 + 47) >> 5;
  for (int kb = 0; kb < nkb; ++kb) {
    const int key0 = kb * 32;
    const bf16* kbA = Kmat + (size_t)(key0 + r15) * DD + hh * DKK + half * 8;
    const bf16* kbB = kbA + (size_t)16 * DD;
    f32x8 c0 = {0.f,0.f,0.f,0.f,0.f,0.f,0.f,0.f};
    f32x8 c1 = c0;
    c0 = wmma_bf16(aq0, load_fragB(kbA),      c0);
    c0 = wmma_bf16(aq1, load_fragB(kbA + 32), c0);
    c1 = wmma_bf16(aq0, load_fragB(kbB),      c1);
    c1 = wmma_bf16(aq1, load_fragB(kbB + 32), c1);

#pragma unroll
    for (int r = 0; r < 8; ++r) {
      const int row = q0 + r + half * 8;
      float v0 = (key0 + r15      <= row) ? c0[r] * 0.125f : -1e30f;
      float v1 = (key0 + 16 + r15 <= row) ? c1[r] * 0.125f : -1e30f;
      float mt = fmaxf(v0, v1);
      mt = fmaxf(mt, __shfl_xor(mt, 1, 32));
      mt = fmaxf(mt, __shfl_xor(mt, 2, 32));
      mt = fmaxf(mt, __shfl_xor(mt, 4, 32));
      mt = fmaxf(mt, __shfl_xor(mt, 8, 32));
      const float mn = fmaxf(mrow[r], mt);
      const float p0 = __expf(v0 - mn);
      const float p1 = __expf(v1 - mn);
      float rs = p0 + p1;
      rs += __shfl_xor(rs, 1, 32);
      rs += __shfl_xor(rs, 2, 32);
      rs += __shfl_xor(rs, 4, 32);
      rs += __shfl_xor(rs, 8, 32);
      const float sca = __expf(mrow[r] - mn);
      lrow[r] = lrow[r] * sca + rs;
      mrow[r] = mn;
      o0[r] *= sca; o1[r] *= sca; o2[r] *= sca; o3[r] *= sca;
      const int prow = r + half * 8;
      pTile[prow * 32 + r15]      = (bf16)p0;
      pTile[prow * 32 + 16 + r15] = (bf16)p1;
    }
    __syncthreads();

    const bf16x16 pf = load_fragA(pTile + r15 * 32 + half * 8);
    const bf16* vbase = Vt + (size_t)(hh * DKK + r15) * SQ + key0 + half * 8;
    o0 = wmma_bf16(pf, load_fragB(vbase),                   o0);
    o1 = wmma_bf16(pf, load_fragB(vbase + (size_t)16 * SQ), o1);
    o2 = wmma_bf16(pf, load_fragB(vbase + (size_t)32 * SQ), o2);
    o3 = wmma_bf16(pf, load_fragB(vbase + (size_t)48 * SQ), o3);
    __syncthreads();
  }

#pragma unroll
  for (int r = 0; r < 8; ++r) {
    const float inv = 1.0f / lrow[r];
    const size_t row = (size_t)(q0 + r + half * 8);
    const int colb = hh * DKK + r15;
    O[row * DD + colb +  0] = (bf16)(o0[r] * inv);
    O[row * DD + colb + 16] = (bf16)(o1[r] * inv);
    O[row * DD + colb + 32] = (bf16)(o2[r] * inv);
    O[row * DD + colb + 48] = (bf16)(o3[r] * inv);
  }
}

// ---------------------------------------------------------------------------
// out = resid + LayerNorm(inp)*g + b ; also writes bf16 copy. One block per row.
// ---------------------------------------------------------------------------
__global__ __launch_bounds__(256) void ln_add_kernel(
    const float* __restrict__ inp, const float* __restrict__ resid,
    const float* __restrict__ g, const float* __restrict__ b,
    float* __restrict__ outF, bf16* __restrict__ outB)
{
  const int row = blockIdx.x, tid = threadIdx.x;
  const float* x = inp + (size_t)row * DD;
  __shared__ float red[256];

  float s = 0.f;
  for (int d = tid; d < DD; d += 256) s += x[d];
  red[tid] = s; __syncthreads();
  for (int off = 128; off > 0; off >>= 1) {
    if (tid < off) red[tid] += red[tid + off];
    __syncthreads();
  }
  const float mu = red[0] * (1.0f / DD);
  __syncthreads();

  float vs = 0.f;
  for (int d = tid; d < DD; d += 256) { float t = x[d] - mu; vs += t * t; }
  red[tid] = vs; __syncthreads();
  for (int off = 128; off > 0; off >>= 1) {
    if (tid < off) red[tid] += red[tid + off];
    __syncthreads();
  }
  const float rstd = rsqrtf(red[0] * (1.0f / DD) + 1e-5f);

  for (int d = tid; d < DD; d += 256) {
    const size_t i = (size_t)row * DD + d;
    const float v = resid[i] + (x[d] - mu) * rstd * g[d] + b[d];
    outF[i] = v;
    outB[i] = (bf16)v;
  }
}

// ---------------------------------------------------------------------------
// h = emb[x] + pos  (fp32 + bf16 copies)
// ---------------------------------------------------------------------------
__global__ __launch_bounds__(256) void embed_kernel(
    const int* __restrict__ x, const float* __restrict__ emb,
    const float* __restrict__ pos, float* __restrict__ hF, bf16* __restrict__ hB)
{
  const int i = blockIdx.x * 256 + threadIdx.x;
  if (i >= SQ * DD) return;
  const int s = i / DD, d = i - s * DD;
  const float v = emb[(size_t)x[s] * DD + d] + pos[i];
  hF[i] = v;
  hB[i] = (bf16)v;
}

// ---------------------------------------------------------------------------
// Wt[n,k](bf16) = W[k,n](f32)   (weight transpose + downconvert)
// ---------------------------------------------------------------------------
__global__ __launch_bounds__(256) void wt_bf16_kernel(
    const float* __restrict__ W, bf16* __restrict__ Wt, int K, int N)
{
  const int i = blockIdx.x * 256 + threadIdx.x;
  if (i >= K * N) return;
  const int n = i / K, k = i - n * K;
  __builtin_prefetch(W + (size_t)(k + 8) * N + n, 0, 1);  // global_prefetch_b8
  Wt[i] = (bf16)W[(size_t)k * N + n];
}

// ---------------------------------------------------------------------------
// Host orchestration
// ---------------------------------------------------------------------------
extern "C" void kernel_launch(void* const* d_in, const int* in_sizes, int n_in,
                              void* d_out, int out_size, void* d_ws, size_t ws_size,
                              hipStream_t stream)
{
  (void)in_sizes; (void)n_in; (void)out_size; (void)ws_size;

  const int*   x     = (const int*)  d_in[0];
  const float* emb   = (const float*)d_in[1];
  const float* pos   = (const float*)d_in[2];
  const float* Wq    = (const float*)d_in[3];
  const float* bq    = (const float*)d_in[4];
  const float* Wk    = (const float*)d_in[5];
  const float* bk    = (const float*)d_in[6];
  const float* Wv    = (const float*)d_in[7];
  const float* bv    = (const float*)d_in[8];
  const float* Wo    = (const float*)d_in[9];
  const float* ln1g  = (const float*)d_in[10];
  const float* ln1b  = (const float*)d_in[11];
  const float* W1    = (const float*)d_in[12];
  const float* b1    = (const float*)d_in[13];
  const float* W2    = (const float*)d_in[14];
  const float* b2    = (const float*)d_in[15];
  const float* ln2g  = (const float*)d_in[16];
  const float* ln2b  = (const float*)d_in[17];
  const float* headw = (const float*)d_in[18];
  const float* headb = (const float*)d_in[19];

  size_t off = 0;
  auto alloc = [&](size_t bytes) -> void* {
    size_t a = (off + 255) & ~(size_t)255;
    off = a + bytes;
    return (void*)((char*)d_ws + a);
  };

  float* hF   = (float*)alloc((size_t)SQ * DD * 4);
  bf16*  hB   = (bf16*) alloc((size_t)SQ * DD * 2);
  bf16*  qB   = (bf16*) alloc((size_t)SQ * DD * 2);
  bf16*  kB   = (bf16*) alloc((size_t)SQ * DD * 2);
  bf16*  vtB  = (bf16*) alloc((size_t)DD * SQ * 2);   // [H*DK, S]
  bf16*  aoB  = (bf16*) alloc((size_t)SQ * DD * 2);
  float* aF   = (float*)alloc((size_t)SQ * DD * 4);
  float* uF   = (float*)alloc((size_t)SQ * DD * 4);
  bf16*  uB   = (bf16*) alloc((size_t)SQ * DD * 2);
  bf16*  m1B  = (bf16*) alloc((size_t)SQ * DHH * 2);
  float* m2F  = (float*)alloc((size_t)SQ * DD * 4);
  bf16*  wqT  = (bf16*) alloc((size_t)DD * DD * 2);
  bf16*  wkT  = (bf16*) alloc((size_t)DD * DD * 2);
  bf16*  wvT  = (bf16*) alloc((size_t)DD * DD * 2);
  bf16*  woT  = (bf16*) alloc((size_t)DD * DD * 2);
  bf16*  w1T  = (bf16*) alloc((size_t)DHH * DD * 2);
  bf16*  w2T  = (bf16*) alloc((size_t)DD * DHH * 2);
  bf16*  hdT  = (bf16*) alloc((size_t)VVN * DD * 2);

  auto transpose = [&](const float* W, bf16* Wt, int K, int N) {
    const int total = K * N;
    wt_bf16_kernel<<<dim3((total + 255) / 256), dim3(256), 0, stream>>>(W, Wt, K, N);
  };

  embed_kernel<<<dim3((SQ * DD) / 256), dim3(256), 0, stream>>>(x, emb, pos, hF, hB);
  transpose(headw, hdT, DD, VVN);

  const dim3 blk(256);
  const dim3 gD (DD  / 128, SQ / 128);   // N=1024
  const dim3 gDH(DHH / 128, SQ / 128);   // N=4096
  const dim3 gV (VVN / 128, SQ / 128);   // N=256

  for (int i = 0; i < LLN; ++i) {
    transpose(Wq + (size_t)i * DD * DD,  wqT, DD,  DD);
    transpose(Wk + (size_t)i * DD * DD,  wkT, DD,  DD);
    transpose(Wv + (size_t)i * DD * DD,  wvT, DD,  DD);
    transpose(Wo + (size_t)i * DD * DD,  woT, DD,  DD);
    transpose(W1 + (size_t)i * DD * DHH, w1T, DD,  DHH);
    transpose(W2 + (size_t)i * DHH * DD, w2T, DHH, DD);

    gemm_wmma_kernel<0, 1><<<gD, blk, 0, stream>>>(hB, wqT, bq + (size_t)i * DD,
        nullptr, qB, SQ, DD, DD, DD, DD, DD);
    gemm_wmma_kernel<0, 1><<<gD, blk, 0, stream>>>(hB, wkT, bk + (size_t)i * DD,
        nullptr, kB, SQ, DD, DD, DD, DD, DD);
    gemm_wmma_kernel<0, 2><<<gD, blk, 0, stream>>>(hB, wvT, bv + (size_t)i * DD,
        nullptr, vtB, SQ, DD, DD, DD, DD, SQ);

    attention_kernel<<<dim3(SQ / 16, HHN), dim3(32), 0, stream>>>(qB, kB, vtB, aoB);

    gemm_wmma_kernel<0, 0><<<gD, blk, 0, stream>>>(aoB, woT, nullptr,
        aF, nullptr, SQ, DD, DD, DD, DD, DD);
    ln_add_kernel<<<dim3(SQ), blk, 0, stream>>>(aF, hF,
        ln1g + (size_t)i * DD, ln1b + (size_t)i * DD, uF, uB);
    gemm_wmma_kernel<1, 1><<<gDH, blk, 0, stream>>>(uB, w1T, b1 + (size_t)i * DHH,
        nullptr, m1B, SQ, DHH, DD, DD, DD, DHH);
    gemm_wmma_kernel<0, 0><<<gD, blk, 0, stream>>>(m1B, w2T, b2 + (size_t)i * DD,
        m2F, nullptr, SQ, DD, DHH, DHH, DHH, DD);
    ln_add_kernel<<<dim3(SQ), blk, 0, stream>>>(m2F, hF,
        ln2g + (size_t)i * DD, ln2b + (size_t)i * DD, hF, hB);
  }

  gemm_wmma_kernel<0, 0><<<gV, blk, 0, stream>>>(hB, hdT, headb,
      (float*)d_out, nullptr, SQ, VVN, DD, DD, DD, VVN);
}
